// GraphSage_79061757984814
// MI455X (gfx1250) — matrix operand order repaired and checked by reading, not compiled, not run on hardware
//
#include <hip/hip_runtime.h>
#include <math.h>

// ---------------------------------------------------------------------------
// GraphSAGE forward for MI455X (gfx1250, wave32).
// Heavy op: a_hat @ h  (a: 12000x12000 f32, binary + exact self loops).
// Bandwidth-bound on reading `a` (576MB/layer, ~74us total at 23.3 TB/s);
// bf16 WMMA keeps compute under the roofline (layer1 needs ~1.5 PF/s at
// 23.3 TB/s; v_wmma_f32_16x16x32_bf16 >> that, f32 16x16x4 WMMA would not
// be). Binary a -> bf16 is exact; h rounded to bf16, f32 accumulation.
// Row degree fused via WMMA against an all-ones B matrix.
// All global indices fit u32 (a = 144M elems) -> SADDR+voffset addressing.
// ---------------------------------------------------------------------------

typedef unsigned short u16;
typedef unsigned int   u32;
typedef __attribute__((ext_vector_type(16))) __bf16 v16bf;
typedef __attribute__((ext_vector_type(8)))  float  v8f;
typedef __attribute__((ext_vector_type(8)))  unsigned int v8u;

__device__ __forceinline__ u16 f2bf(float f) {
  return __builtin_bit_cast(u16, (__bf16)f);   // hw cvt, RNE
}
__device__ __forceinline__ float bf2f(u16 h) {
  return __uint_as_float(((u32)h) << 16);
}

__device__ __forceinline__ v8f v8f_zero() {
  v8f z;
#pragma unroll
  for (int e = 0; e < 8; ++e) z[e] = 0.0f;
  return z;
}

__device__ __forceinline__ v16bf pack8(uint4 lo, uint4 hi) {
  v8u u;
  u[0] = lo.x; u[1] = lo.y; u[2] = lo.z; u[3] = lo.w;
  u[4] = hi.x; u[5] = hi.y; u[6] = hi.z; u[7] = hi.w;
  return __builtin_bit_cast(v16bf, u);
}

__device__ __forceinline__ v8f wmma_bf16(v16bf a, v16bf b, v8f c) {
  // v_wmma_f32_16x16x32_bf16
  return __builtin_amdgcn_wmma_f32_16x16x32_bf16(false, a, false, b,
                                                 (short)0, c, false, false);
}

// A-operand (16x32 bf16) from LDS row-major tile. rowp = row start.
// halfSel = lane>>4. Layout: lanes<16 hold K[kc+0..15];
// lanes>=16 hold K[kc+16..31]; 2 bf16 per VGPR.
__device__ __forceinline__ v16bf load_a_lds(const u16* rowp, int kc, int halfSel) {
  const uint4* p = (const uint4*)(rowp + kc + halfSel * 16);
  return pack8(p[0], p[1]);   // +0..15 (ushorts)
}

// B-operand (32x16 bf16) from LDS column-major (hT[c][k]) tile. colp = col start.
// lanes<16: K[kc+0..15]; lanes>=16: K[kc+16..31]; contiguous 32B.
__device__ __forceinline__ v16bf load_b_lds(const u16* colp, int kc, int khalf) {
  const uint4* p = (const uint4*)(colp + kc + khalf);
  return pack8(p[0], p[1]);
}

// ---------------------------------------------------------------------------
// One SAGE layer: agg = (a_hat @ h)/deg ; o = [h,agg]@W + b ; l2norm ; tanh.
// WIN = input feature width (128 for layer 1, 32 after).
// Block = 128 threads (4 waves), 64 output rows/block, K tile = 64.
// ---------------------------------------------------------------------------
template <int WIN>
__global__ __launch_bounds__(128)
void sage_layer_kernel(const float* __restrict__ a,
                       const u16*  __restrict__ h_in,   // [n][WIN] bf16
                       const u16*  __restrict__ pw,     // packed B-layout weights
                       const float* __restrict__ bias,  // [32]
                       u16*        __restrict__ h_out,  // [n][32] bf16
                       int n)
{
  constexpr int KT  = 64;
  constexpr int SRA = KT + 8;        // LDS row stride (ushorts), bank-spread
  constexpr int NT  = WIN / 16;      // B column tiles in main loop
  constexpr int SAG = WIN + 8;       // agg LDS row stride
  constexpr int CW  = WIN / 8;       // 8-wide col groups in hT loader

  __shared__ u16 sA[64 * SRA];
  __shared__ u16 sH[WIN * SRA];            // transposed: sH[c][k]
  __shared__ u16 sAgg[4 * 16 * SAG];       // per-wave agg tile (bf16)

  const int tid     = threadIdx.x;
  const int lane    = tid & 31;
  const int wave    = tid >> 5;
  const int halfSel = lane >> 4;
  const int l15     = lane & 15;
  const int rowBase = blockIdx.x * 64;
  const u32 un      = (u32)n;

  v8f accT[NT];
#pragma unroll
  for (int t = 0; t < NT; ++t) accT[t] = v8f_zero();
  v8f accD = v8f_zero();                   // degree accumulator (A x ones)

  v8u ou;
#pragma unroll
  for (int e = 0; e < 8; ++e) ou[e] = 0x3F803F80u;  // bf16(1.0) x2
  const v16bf ONES = __builtin_bit_cast(v16bf, ou);

  const int nK = (n + KT - 1) / KT;
  const bool rowsFull = (rowBase + 64 <= n);

  for (int kb = 0; kb < nK; ++kb) {
    const int kBase = kb * KT;
    // block-uniform predicates: fast path for interior tiles
    const bool kFull   = (kBase + KT <= n);
    const bool diagHit = (kBase < rowBase + 64) && (rowBase < kBase + KT);

    // ---- stage a tile (64x64 f32 -> bf16, with self-loop fixup) ----
    if (rowsFull && kFull) {
#pragma unroll
      for (int i = 0; i < 8; ++i) {
        int idx = tid + 128 * i;           // 0..1023
        int row = idx >> 4;                // 0..63
        int col = (idx & 15) * 4;
        union { float4 v; float f[4]; } u;
        u.v = *(const float4*)(a + ((u32)(rowBase + row) * un + (u32)(kBase + col)));
        if (diagHit) {                     // uniform branch, rare
          int gr = rowBase + row, gc = kBase + col;
#pragma unroll
          for (int j = 0; j < 4; ++j)
            if (gc + j == gr) u.f[j] += (fabsf(u.f[j]) < 1e-6f) ? 1.0f : 0.0f;
        }
        u32 w0 = (u32)f2bf(u.f[0]) | ((u32)f2bf(u.f[1]) << 16);
        u32 w1 = (u32)f2bf(u.f[2]) | ((u32)f2bf(u.f[3]) << 16);
        *(uint2*)(&sA[row * SRA + col]) = make_uint2(w0, w1);
      }
    } else {
#pragma unroll
      for (int i = 0; i < 8; ++i) {
        int idx = tid + 128 * i;
        int row = idx >> 4;
        int col = (idx & 15) * 4;
        int gr = rowBase + row, gc = kBase + col;
        union { float4 v; float f[4]; } u;
        u.v = make_float4(0.f, 0.f, 0.f, 0.f);
        if (gr < n) {
          if (gc + 3 < n) {
            u.v = *(const float4*)(a + ((u32)gr * un + (u32)gc));
          } else {
#pragma unroll
            for (int j = 0; j < 4; ++j)
              if (gc + j < n) u.f[j] = a[(u32)gr * un + (u32)(gc + j)];
          }
        }
#pragma unroll
        for (int j = 0; j < 4; ++j)
          if (gc + j == gr) u.f[j] += (fabsf(u.f[j]) < 1e-6f) ? 1.0f : 0.0f;
        u32 w0 = (u32)f2bf(u.f[0]) | ((u32)f2bf(u.f[1]) << 16);
        u32 w1 = (u32)f2bf(u.f[2]) | ((u32)f2bf(u.f[3]) << 16);
        *(uint2*)(&sA[row * SRA + col]) = make_uint2(w0, w1);
      }
    }

    // ---- stage h tile transposed: sH[c][k] = h_in[kBase+k][c] ----
    if (kFull) {
#pragma unroll
      for (int i = 0; i < (64 * CW) / 128; ++i) {
        int lid = tid + 128 * i;
        int kr  = lid / CW;
        int c0  = (lid % CW) * 8;
        union { uint4 v; u16 s[8]; } u;
        u.v = *(const uint4*)(h_in + ((u32)(kBase + kr) * (u32)WIN + (u32)c0));
#pragma unroll
        for (int j = 0; j < 8; ++j) sH[(c0 + j) * SRA + kr] = u.s[j];
      }
    } else {
#pragma unroll
      for (int i = 0; i < (64 * CW) / 128; ++i) {
        int lid = tid + 128 * i;
        int kr  = lid / CW;
        int c0  = (lid % CW) * 8;
        int gk  = kBase + kr;
        union { uint4 v; u16 s[8]; } u;
        u.v = make_uint4(0u, 0u, 0u, 0u);
        if (gk < n) u.v = *(const uint4*)(h_in + ((u32)gk * (u32)WIN + (u32)c0));
#pragma unroll
        for (int j = 0; j < 8; ++j) sH[(c0 + j) * SRA + kr] = u.s[j];
      }
    }

    // prefetch next a tile (covers the 16KB tile at 128B/thread)
    if (kb + 1 < nK) {
      int prow = tid >> 1, pcol = (tid & 1) * 32;
      int gr = rowBase + prow, gc = kBase + KT + pcol;
      if (gr < n && gc < n)
        __builtin_prefetch(a + ((u32)gr * un + (u32)gc), 0, 1);
    }

    __syncthreads();

    // ---- WMMA: 16 rows per wave, all WIN cols, plus degree ----
    const u16* arow = &sA[(wave * 16 + l15) * SRA];
#pragma unroll
    for (int kc = 0; kc < KT; kc += 32) {
      v16bf A = load_a_lds(arow, kc, halfSel);
      accD = wmma_bf16(A, ONES, accD);     // fused row-degree
#pragma unroll
      for (int t = 0; t < NT; ++t) {
        const u16* colp = &sH[(16 * t + l15) * SRA];
        v16bf B = load_b_lds(colp, kc, halfSel ? 16 : 0);
        accT[t] = wmma_bf16(A, B, accT[t]);
      }
    }
    __syncthreads();
  }

  // ---- epilogue: agg = acc * (1/deg) -> LDS (A-layout source) ----
  float rdeg[8];
#pragma unroll
  for (int vj = 0; vj < 8; ++vj) rdeg[vj] = 1.0f / accD[vj];

  u16* aggp = &sAgg[wave * 16 * SAG];
#pragma unroll
  for (int t = 0; t < NT; ++t) {
#pragma unroll
    for (int vj = 0; vj < 8; ++vj) {
      int row = vj + 8 * halfSel;          // C-layout row
      int col = 16 * t + l15;              // C-layout col
      aggp[row * SAG + col] = f2bf(accT[t][vj] * rdeg[vj]);
    }
  }

  // ---- out = [h, agg] @ W + b via WMMA (weights pre-packed B-layout) ----
  v8f accO[2] = { v8f_zero(), v8f_zero() };
  const int row_g_a = rowBase + wave * 16 + l15;   // A-operand row for this lane

  // part 0: h rows straight from global (bf16, row-major = A layout segments)
#pragma unroll
  for (int kc = 0; kc < WIN; kc += 32) {
    uint4 lo = make_uint4(0u,0u,0u,0u), hi = lo;
    if (row_g_a < n) {
      const u16* hp = h_in + ((u32)row_g_a * (u32)WIN + (u32)(kc + halfSel * 16));
      lo = *(const uint4*)hp;
      hi = *(const uint4*)(hp + 8);
    }
    v16bf A = pack8(lo, hi);
#pragma unroll
    for (int t = 0; t < 2; ++t) {
      int tileIdx = (kc / 32) * 2 + t;
      v8u bu = *(const v8u*)(pw + (u32)((tileIdx * 32 + lane) * 16));
      accO[t] = wmma_bf16(A, __builtin_bit_cast(v16bf, bu), accO[t]);
    }
  }

  __syncthreads();   // make sAgg writes visible across lanes

  // part 1: agg from LDS
#pragma unroll
  for (int kc = 0; kc < WIN; kc += 32) {
    const u16* ap = aggp + l15 * SAG;
    v16bf A = load_a_lds(ap, kc, halfSel);
#pragma unroll
    for (int t = 0; t < 2; ++t) {
      int tileIdx = (WIN / 32 + kc / 32) * 2 + t;
      v8u bu = *(const v8u*)(pw + (u32)((tileIdx * 32 + lane) * 16));
      accO[t] = wmma_bf16(A, __builtin_bit_cast(v16bf, bu), accO[t]);
    }
  }

  // ---- bias + row l2-normalize + tanh + store bf16 ----
  const float b0 = bias[l15];
  const float b1v = bias[16 + l15];
#pragma unroll
  for (int vj = 0; vj < 8; ++vj) {
    float o0 = accO[0][vj] + b0;
    float o1 = accO[1][vj] + b1v;
    float ss = o0 * o0 + o1 * o1;
    ss += __shfl_xor(ss, 1);
    ss += __shfl_xor(ss, 2);
    ss += __shfl_xor(ss, 4);
    ss += __shfl_xor(ss, 8);               // reduce over 16-lane half (one row)
    float sc = rsqrtf(fmaxf(ss, 1e-12f));
    o0 = tanhf(o0 * sc);
    o1 = tanhf(o1 * sc);
    int row = rowBase + wave * 16 + vj + 8 * halfSel;
    if (row < n) {
      h_out[(u32)row * 32u + (u32)l15]        = f2bf(o0);
      h_out[(u32)row * 32u + 16u + (u32)l15]  = f2bf(o1);
    }
  }
}

// ---------------------------------------------------------------------------
// Helpers: f32 -> bf16 conversion, weight packing into WMMA B-operand layout.
// ---------------------------------------------------------------------------
__global__ void cvt_f32_bf16(const float* __restrict__ src,
                             u16* __restrict__ dst, int nElem) {
  int i = blockIdx.x * blockDim.x + threadIdx.x;
  if (i < nElem) dst[i] = f2bf(src[i]);
}

// W: [2*win][32] row-major f32 -> pw tiles of (kcI, t): lane l, elem e holds
// W[kcI*32 + (l<16?0:16) + e][16*t + (l&15)] as bf16 (contiguous per lane).
__global__ void pack_weights(const float* __restrict__ W,
                             u16* __restrict__ pw, int win) {
  int nTiles = (2 * win / 32) * 2;
  int total = nTiles * 32 * 16;
  for (int i = blockIdx.x * blockDim.x + threadIdx.x; i < total;
       i += blockDim.x * gridDim.x) {
    int e    = i & 15;
    int l    = (i >> 4) & 31;
    int tile = i >> 9;
    int kcI = tile >> 1, t = tile & 1;
    int col = 16 * t + (l & 15);
    int k   = kcI * 32 + ((l < 16) ? 0 : 16) + e;
    pw[i] = f2bf(W[k * 32 + col]);
  }
}

// ---------------------------------------------------------------------------
// Global sum pool + 2-layer MLP head (tiny; single block).
// ---------------------------------------------------------------------------
__global__ __launch_bounds__(256)
void pool_mlp(const u16* __restrict__ h3, const float* __restrict__ Wf1,
              const float* __restrict__ bf1, const float* __restrict__ Wf2,
              const float* __restrict__ bf2, float* __restrict__ out, int n) {
  __shared__ float sp[8][32];
  __shared__ float pcol[32];
  __shared__ float th[64];
  int tid = threadIdx.x;
  int c = tid & 31, g = tid >> 5;
  float s = 0.f;
  for (int i = g; i < n; i += 8) s += bf2f(h3[(u32)i * 32u + (u32)c]);
  sp[g][c] = s;
  __syncthreads();
  if (tid < 32) {
    float p = 0.f;
    for (int g2 = 0; g2 < 8; ++g2) p += sp[g2][tid];
    pcol[tid] = p;
  }
  __syncthreads();
  if (tid < 64) {
    float v = bf1[tid];
    for (int cc = 0; cc < 32; ++cc) v += pcol[cc] * Wf1[cc * 64 + tid];
    th[tid] = tanhf(v);
  }
  __syncthreads();
  if (tid == 0) {
    float r = bf2[0];
    for (int j = 0; j < 64; ++j) r += th[j] * Wf2[j];
    out[0] = r;
  }
}

// ---------------------------------------------------------------------------
extern "C" void kernel_launch(void* const* d_in, const int* in_sizes, int n_in,
                              void* d_out, int out_size, void* d_ws, size_t ws_size,
                              hipStream_t stream) {
  (void)n_in; (void)out_size; (void)ws_size;
  const float* x   = (const float*)d_in[0];
  const float* a   = (const float*)d_in[1];
  const float* W1  = (const float*)d_in[2];
  const float* b1  = (const float*)d_in[3];
  const float* W2  = (const float*)d_in[4];
  const float* b2  = (const float*)d_in[5];
  const float* W3  = (const float*)d_in[6];
  const float* b3  = (const float*)d_in[7];
  const float* Wf1 = (const float*)d_in[8];
  const float* bf1 = (const float*)d_in[9];
  const float* Wf2 = (const float*)d_in[10];
  const float* bf2 = (const float*)d_in[11];
  float* out = (float*)d_out;

  const int F = 128, H = 32;
  const int n = in_sizes[0] / F;           // 12000

  char* ws = (char*)d_ws;
  size_t off = 0;
  auto walloc = [&](size_t bytes) -> void* {
    void* p = ws + off;
    off = (off + bytes + 255) & ~(size_t)255;
    return p;
  };
  u16* xb  = (u16*)walloc((size_t)n * F * 2);
  u16* h1  = (u16*)walloc((size_t)n * H * 2);
  u16* h2  = (u16*)walloc((size_t)n * H * 2);
  u16* h3  = (u16*)walloc((size_t)n * H * 2);
  u16* pw1 = (u16*)walloc((size_t)2 * F * 32 * 2);       // 16 KB
  u16* pw2 = (u16*)walloc((size_t)2 * (2 * H) * 16 * 2); // 4 KB
  u16* pw3 = (u16*)walloc((size_t)2 * (2 * H) * 16 * 2); // 4 KB

  cvt_f32_bf16<<<(n * F + 255) / 256, 256, 0, stream>>>(x, xb, n * F);
  pack_weights<<<16, 256, 0, stream>>>(W1, pw1, F);   // [256][32]
  pack_weights<<<4, 256, 0, stream>>>(W2, pw2, H);    // [64][32]
  pack_weights<<<4, 256, 0, stream>>>(W3, pw3, H);

  int grid = (n + 63) / 64;
  sage_layer_kernel<128><<<grid, 128, 0, stream>>>(a, xb, pw1, b1, h1, n);
  sage_layer_kernel<32><<<grid, 128, 0, stream>>>(a, h1, pw2, b2, h2, n);
  sage_layer_kernel<32><<<grid, 128, 0, stream>>>(a, h2, pw3, b3, h3, n);

  pool_mlp<<<1, 256, 0, stream>>>(h3, Wf1, bf1, Wf2, bf2, out, n);
}
